// Model_70111046140521
// MI455X (gfx1250) — matrix operand order repaired and checked by reading, not compiled, not run on hardware
//
#include <hip/hip_runtime.h>
#include <hip/hip_bf16.h>

// ---------------------------------------------------------------------------
// S4 world-model forward for MI455X (gfx1250, wave32, WMMA, TDM).
// Recurrence is re-blocked into 16-step chunks so the sequential chain is
// 128 steps/layer (LDS matvec, pitch-129 to kill bank conflicts) and the
// bulk of the FLOPs become v_wmma_f32_16x16x32_f16 GEMMs. The per-block
// P/M staging into LDS uses the Tensor Data Mover with descriptor padding
// generating the bank-conflict-free pitch directly.
// ---------------------------------------------------------------------------

#define EPSc 1e-5f
constexpr int Bc = 2, Lc = 2048, Sc = 32, Ac = 8, Hc = 128, Nc = 64, NLc = 4;
constexpr int RS = 2 * Nc;        // 128: real-embedded complex state
constexpr int CH = 16;            // chunk length
constexpr int NC = Lc / CH;       // 128 chunks
constexpr int LH = NLc * Hc;      // 512 (layer,channel) pairs
constexpr int ROWS = Bc * Lc;     // 4096

typedef __attribute__((ext_vector_type(16))) _Float16 v16h;
typedef __attribute__((ext_vector_type(8)))  float    v8f;

// ---- Tensor Data Mover helper (gfx1250 only; 6-arg clang-23 builtin) ------
#if defined(__gfx1250__) && __has_builtin(__builtin_amdgcn_tensor_load_to_lds)
#define HAVE_TDM 1
typedef unsigned int u32x4 __attribute__((ext_vector_type(4)));
typedef int          i32x8 __attribute__((ext_vector_type(8)));
typedef int          i32x4 __attribute__((ext_vector_type(4)));

// Async-load a 2D f32 tile (dim0 x dim1, row stride stride0 elements) from
// global to LDS at byte offset lds_off. pad_code inserts 1 DWORD of LDS
// padding every (2<<pad_code) DWORDs -> builds pitched LDS rows in-flight.
__device__ __forceinline__ void tdm_load_2d(const void* gptr, unsigned lds_off,
                                            unsigned dim0, unsigned dim1,
                                            unsigned stride0, unsigned pad_code) {
  unsigned long long ga = (unsigned long long)gptr;
  // D# group0: count=1 | lds_addr | global_addr[56:0] | type=2
  u32x4 g0 = { 1u, lds_off, (unsigned)ga,
               ((unsigned)((ga >> 32) & 0x01ffffffu)) | 0x80000000u };
  // D# group1: data_size=4B, pad_enable, pad_interval=pad_code, pad_amount=1dw,
  //            tensor_dim0/1, tile_dim0/1 (= full tensor), dim0 stride.
  unsigned w0 = (2u << 16) | (1u << 20) | (pad_code << 22);
  i32x8 g1 = { (int)w0,
               (int)(dim0 << 16),                      // tensor_dim0[15:0]
               (int)((dim0 >> 16) | (dim1 << 16)),     // dim0 hi | tensor_dim1 lo
               (int)((dim1 >> 16) | (dim0 << 16)),     // dim1 hi | tile_dim0
               (int)(dim1 & 0xffffu),                  // tile_dim1 (tile_dim2=0)
               (int)stride0,                           // tensor_dim0_stride lo
               0, 0 };
  i32x4 gz4 = { 0, 0, 0, 0 };                          // 2D: groups 2/3 unused
  i32x8 gz8 = { 0, 0, 0, 0, 0, 0, 0, 0 };              // extra group (clang-23)
  __builtin_amdgcn_tensor_load_to_lds(g0, g1, gz4, gz4, gz8, 0);
}
#endif

// ---- WMMA fragment helpers (wave32 layouts per CDNA5 ISA 7.12.2) ----------

// A fragment: 16x32 f16. Element (m,k) read from f32 src at p[m*ldm + k*ldk].
// lane<16: row=lane, K = c + (c>=8?8:0); lane>=16: same row, K += 8.
__device__ __forceinline__ v16h load_a(const float* p, int ldm, int ldk, int kvalid) {
  int lane = threadIdx.x & 31;
  int m = lane & 15;
  int hi = lane >> 4;
  v16h a;
#pragma unroll
  for (int c = 0; c < 16; ++c) {
    int k = c + (c >= 8 ? 8 : 0) + (hi ? 8 : 0);
    float v = (k < kvalid) ? p[m * ldm + k * ldk] : 0.0f;
    a[c] = (_Float16)v;
  }
  return a;
}

// B fragment: 32x16 f16. Element (k,n) at p[k*ldk + n*ldn]. col=lane&15, K as in load_a.
__device__ __forceinline__ v16h load_b(const float* p, int ldk, int ldn, int kvalid) {
  int lane = threadIdx.x & 31;
  int n = lane & 15;
  int hi = lane >> 4;
  v16h b;
#pragma unroll
  for (int c = 0; c < 16; ++c) {
    int k = c + (c >= 8 ? 8 : 0) + (hi ? 8 : 0);
    float v = (k < kvalid) ? p[k * ldk + n * ldn] : 0.0f;
    b[c] = (_Float16)v;
  }
  return b;
}

__device__ __forceinline__ v8f wmma32(v16h a, v16h b, v8f c) {
  // 8 args: (neg_a, A, neg_b, B, c_mod, C, reuse_a, reuse_b)
  return __builtin_amdgcn_wmma_f32_16x16x32_f16(false, a, false, b, (short)0, c,
                                                false, false);
}

// C/D 16x16 f32: VGPR r -> M = r + 8*(lane>=16); N = lane&15.
__device__ __forceinline__ void store_cd(float* p, int ld, v8f acc) {
  int lane = threadIdx.x & 31;
  int n = lane & 15;
  int mb = (lane >> 4) * 8;
#pragma unroll
  for (int r = 0; r < 8; ++r) p[(mb + r) * ld + n] = acc[r];
}

// ---- Kernels ---------------------------------------------------------------

// x[row, col] = concat(state,action)[row,:] . enc_w[col,:] + enc_b[col]
__global__ void k_encode(const float* __restrict__ st, const float* __restrict__ ac,
                         const float* __restrict__ ew, const float* __restrict__ eb,
                         float* __restrict__ x) {
  int idx = blockIdx.x * blockDim.x + threadIdx.x;
  if (idx >= ROWS * Hc) return;
  int row = idx / Hc, col = idx % Hc;
  const float* w = ew + col * (Sc + Ac);
  const float* s = st + (size_t)row * Sc;
  const float* a = ac + (size_t)row * Ac;
  float acc = eb[col];
#pragma unroll
  for (int k = 0; k < Sc; ++k) acc += s[k] * w[k];
#pragma unroll
  for (int k = 0; k < Ac; ++k) acc += a[k] * w[Sc + k];
  x[idx] = acc;
}

// LayerNorm over H; writes xn [B,L,H] and transposed copy uT [B,H,L].
__global__ void k_ln(const float* __restrict__ x, const float* __restrict__ nw,
                     const float* __restrict__ nb, int layer,
                     float* __restrict__ xn, float* __restrict__ uT) {
  __shared__ float s1[Hc], s2[Hc];
  int row = blockIdx.x, tid = threadIdx.x;
  float v = x[(size_t)row * Hc + tid];
  s1[tid] = v; s2[tid] = v * v;
  __syncthreads();
  for (int off = Hc / 2; off > 0; off >>= 1) {
    if (tid < off) { s1[tid] += s1[tid + off]; s2[tid] += s2[tid + off]; }
    __syncthreads();
  }
  float mu = s1[0] * (1.0f / Hc);
  float var = s2[0] * (1.0f / Hc) - mu * mu;
  float o = (v - mu) * rsqrtf(var + EPSc) * nw[layer * Hc + tid] + nb[layer * Hc + tid];
  xn[(size_t)row * Hc + tid] = o;
  int b = row / Lc, t = row % Lc;
  uT[((size_t)b * Hc + tid) * Lc + t] = o;
}

// Materialize real embedding R(A) = [[Ar,-Ai],[Ai,Ar]] per (layer,channel).
__global__ void k_embed(const float* __restrict__ Ab, float* __restrict__ RA) {
  int lh = blockIdx.x;
  const float* A = Ab + (size_t)lh * Nc * Nc * 2;   // complex64 interleaved
  float* R = RA + (size_t)lh * RS * RS;
  for (int e = threadIdx.x; e < RS * RS; e += blockDim.x) {
    int i = e / RS, j = e % RS;
    int n = i & (Nc - 1), m = j & (Nc - 1);
    float re = A[(n * Nc + m) * 2];
    float im = A[(n * Nc + m) * 2 + 1];
    R[e] = (i < Nc) ? (j < Nc ? re : -im) : (j < Nc ? im : re);
  }
}

// b_hat = [Br;Bi], c_hat = [Cr,-Ci] (so c_hat . [hr;hi] = Re(C h)).
__global__ void k_bc(const float* __restrict__ Bb, const float* __restrict__ Cb,
                     float* __restrict__ Bh, float* __restrict__ Ch) {
  int lh = blockIdx.x, i = threadIdx.x;
  int n = i & (Nc - 1);
  float br = Bb[((size_t)lh * Nc + n) * 2], bi = Bb[((size_t)lh * Nc + n) * 2 + 1];
  float cr = Cb[((size_t)lh * Nc + n) * 2], ci = Cb[((size_t)lh * Nc + n) * 2 + 1];
  Bh[(size_t)lh * RS + i] = (i < Nc) ? br : bi;
  Ch[(size_t)lh * RS + i] = (i < Nc) ? cr : -ci;
}

// Batched square C = A*A over 512 matrices of 128x128 (WMMA, 8 waves/block).
__global__ void k_sq(const float* __restrict__ A, float* __restrict__ C) {
  int wid = threadIdx.x >> 5;
  int bm = blockIdx.x >> 3;
  int tile = (blockIdx.x & 7) * 8 + wid;      // 0..63
  int tm = tile >> 3, tn = tile & 7;
  const float* M = A + (size_t)bm * RS * RS;
  v8f acc = {};
#pragma unroll
  for (int kk = 0; kk < 4; ++kk) {
    v16h af = load_a(M + (size_t)tm * 16 * RS + kk * 32, RS, 1, 32);
    v16h bf = load_b(M + (size_t)kk * 32 * RS + tn * 16, RS, 1, 32);
    acc = wmma32(af, bf, acc);
  }
  store_cd(C + (size_t)bm * RS * RS + (size_t)tm * 16 * RS + tn * 16, RS, acc);
}

// Per (layer,channel): W_hat rows = c_hat R^{t+1}; M_hat col (15-t) = R^t b_hat;
// kappa_t = c_hat . R^t b_hat;  T_hat[t][k] = kappa_{t-k} (k<=t), +D on diag.
__global__ void k_wmk(const float* __restrict__ RA, const float* __restrict__ Bh,
                      const float* __restrict__ Ch, const float* __restrict__ Dp,
                      float* __restrict__ What, float* __restrict__ Mhat,
                      float* __restrict__ That) {
  __shared__ float w[RS], v[RS], red[RS], kap[CH];
  int lh = blockIdx.x, tid = threadIdx.x;
  int layer = lh / Hc, h = lh % Hc;
  const float* R = RA + (size_t)lh * RS * RS;
  float chat = Ch[(size_t)lh * RS + tid];
  w[tid] = chat;
  v[tid] = Bh[(size_t)lh * RS + tid];
  __syncthreads();
  for (int t = 0; t < CH; ++t) {
    // kappa_t = c_hat . v_t
    red[tid] = chat * v[tid];
    __syncthreads();
    for (int off = RS / 2; off > 0; off >>= 1) {
      if (tid < off) red[tid] += red[tid + off];
      __syncthreads();
    }
    if (tid == 0) kap[t] = red[0];
    Mhat[(size_t)lh * RS * CH + tid * CH + (CH - 1 - t)] = v[tid];
    // w <- w R (row * matrix), v <- R v (matrix * col)
    float aw = 0.0f, av = 0.0f;
    for (int m = 0; m < RS; ++m) {
      aw += w[m] * R[(size_t)m * RS + tid];
      av += R[(size_t)tid * RS + m] * v[m];
    }
    __syncthreads();
    w[tid] = aw; v[tid] = av;
    __syncthreads();
    What[(size_t)lh * CH * RS + t * RS + tid] = w[tid];   // c_hat R^{t+1}
  }
  __syncthreads();
  for (int e = tid; e < CH * 32; e += blockDim.x) {
    int t = e >> 5, k = e & 31;
    float val = (k < CH && k <= t) ? kap[t - k] : 0.0f;
    if (k == t) val += Dp[layer * Hc + h];
    That[(size_t)lh * CH * 32 + e] = val;
  }
}

// Sequential chunk scan: h_{c+1} = P16 h_c + M_hat u_c. Stores h_c (chunk-start
// state) into Hmat[(b,h)][state][chunk]. P16 staged into LDS with pitch 129
// (bank = (tid + m) mod 64 -> conflict-free); on gfx1250 the staging is a
// single TDM tensor_load_to_lds per matrix whose pad descriptor creates the
// pitch during the DMA.
__global__ void k_scan(const float* __restrict__ P16, const float* __restrict__ Mhat,
                       const float* __restrict__ xn, int layer,
                       float* __restrict__ Hmat) {
  extern __shared__ float sm[];
  const int PP = RS + 1, MP = CH + 1;
  float* P  = sm;
  float* Ms = P + RS * PP;
  float* hb = Ms + RS * MP;
  float* ub = hb + RS;
  int bh = blockIdx.x;
  int b = bh / Hc, h = bh % Hc;
  int lh = layer * Hc + h;
  int tid = threadIdx.x;                       // 128 threads = 4 waves
  const float* Pg = P16 + (size_t)lh * RS * RS;
  const float* Mg = Mhat + (size_t)lh * RS * CH;
#ifdef HAVE_TDM
  if (tid < 32) {                              // wave 0 issues the DMAs once
    // P: 128x128 f32, pad 1 DWORD every 128 DWORDs -> LDS pitch 129
    tdm_load_2d(Pg, (unsigned)(size_t)P, RS, RS, RS, 6u);
    // M_hat: 128 rows x 16 f32, pad 1 DWORD every 16 DWORDs -> pitch 17
    tdm_load_2d(Mg, (unsigned)(size_t)Ms, CH, RS, CH, 3u);
    __builtin_amdgcn_s_wait_tensorcnt(0);
  }
#else
  for (int e = tid; e < RS * RS; e += blockDim.x) P[(e / RS) * PP + (e % RS)] = Pg[e];
  for (int e = tid; e < RS * CH; e += blockDim.x) Ms[(e / CH) * MP + (e % CH)] = Mg[e];
#endif
  hb[tid] = 0.0f;
  __syncthreads();
  float* Hrow = Hmat + ((size_t)bh * RS + tid) * NC;
  for (int c = 0; c < NC; ++c) {
    if (tid < CH) ub[tid] = xn[((size_t)b * Lc + c * CH + tid) * Hc + h];
    if (c + 1 < NC && tid < CH)
      __builtin_prefetch(&xn[((size_t)b * Lc + (c + 1) * CH + tid) * Hc + h], 0, 0);
    __syncthreads();
    Hrow[c] = hb[tid];                          // state at chunk start
    float acc = 0.0f;
#pragma unroll 8
    for (int m = 0; m < RS; ++m) acc += P[tid * PP + m] * hb[m];
#pragma unroll
    for (int j = 0; j < CH; ++j) acc += Ms[tid * MP + j] * ub[j];
    __syncthreads();
    hb[tid] = acc;
    __syncthreads();
  }
}

// Per (b,h): Y[16 x 128chunks] = W_hat(16x128) H + T_hat(16x16) U; then gelu,
// scattered into yg [B,L,H]. One wave per 16-chunk tile (8 waves/block).
__global__ void k_out(const float* __restrict__ What, const float* __restrict__ That,
                      const float* __restrict__ Hmat, const float* __restrict__ uT,
                      int layer, float* __restrict__ yg) {
  int bh = blockIdx.x;
  int b = bh / Hc, h = bh % Hc;
  int lh = layer * Hc + h;
  int wid = threadIdx.x >> 5;                  // chunk tile 0..7
  const float* Wp = What + (size_t)lh * CH * RS;
  const float* Hp = Hmat + (size_t)bh * RS * NC + wid * CH;
  v8f acc = {};
#pragma unroll
  for (int kk = 0; kk < 4; ++kk) {
    v16h af = load_a(Wp + kk * 32, RS, 1, 32);
    v16h bf = load_b(Hp + (size_t)kk * 32 * NC, NC, 1, 32);
    acc = wmma32(af, bf, acc);
  }
  { // Toeplitz intra-chunk term, K padded 16->32 with zeros
    v16h af = load_a(That + (size_t)lh * CH * 32, 32, 1, 32);
    const float* Up = uT + ((size_t)b * Hc + h) * Lc + wid * CH * CH;
    v16h bf = load_b(Up, 1, CH, CH);           // (k,n) -> u[(tile*16+n)*16+k]
    acc = wmma32(af, bf, acc);
  }
  int lane = threadIdx.x & 31;
  int n = lane & 15, mb = (lane >> 4) * 8;
#pragma unroll
  for (int r = 0; r < 8; ++r) {
    int m = mb + r;
    int c = wid * CH + n;
    int t = c * CH + m;
    float s = acc[r];
    float g = 0.5f * s * (1.0f + tanhf(0.7978845608f * (s + 0.044715f * s * s * s)));
    yg[((size_t)b * Lc + t) * Hc + h] = g;
  }
}

// Gated MLP + residual: x_out = skip + (G@ow^T+ob) * sigmoid(G@o2w^T+o2b).
__global__ void k_gate(const float* __restrict__ yg, const float* __restrict__ ow,
                       const float* __restrict__ ob, const float* __restrict__ o2w,
                       const float* __restrict__ o2b, const float* __restrict__ skip,
                       int layer, float* __restrict__ xo) {
  int r0 = blockIdx.x * 16;
  int c0 = (threadIdx.x >> 5) * 16;
  const float* A  = yg + (size_t)r0 * Hc;
  const float* W1 = ow  + (size_t)layer * Hc * Hc + (size_t)c0 * Hc;
  const float* W2 = o2w + (size_t)layer * Hc * Hc + (size_t)c0 * Hc;
  v8f a1 = {}, a2 = {};
#pragma unroll
  for (int kk = 0; kk < 4; ++kk) {
    v16h af = load_a(A + kk * 32, Hc, 1, 32);
    v16h b1 = load_b(W1 + kk * 32, 1, Hc, 32); // (k,n) -> ow[(c0+n)*H + k]
    v16h b2 = load_b(W2 + kk * 32, 1, Hc, 32);
    a1 = wmma32(af, b1, a1);
    a2 = wmma32(af, b2, a2);
  }
  int lane = threadIdx.x & 31;
  int n = lane & 15, mb = (lane >> 4) * 8;
#pragma unroll
  for (int r = 0; r < 8; ++r) {
    int row = r0 + mb + r, col = c0 + n;
    float u = a1[r] + ob[layer * Hc + col];
    float g = a2[r] + o2b[layer * Hc + col];
    float val = u * (1.0f / (1.0f + expf(-g)));
    xo[(size_t)row * Hc + col] = skip[(size_t)row * Hc + col] + val;
  }
}

// next_state = x @ sdec_w^T + sdec_b  (M=16 rows/block, 2 waves = 32 cols)
__global__ void k_sdec(const float* __restrict__ x, const float* __restrict__ sw,
                       const float* __restrict__ sb, float* __restrict__ out) {
  int r0 = blockIdx.x * 16;
  int c0 = (threadIdx.x >> 5) * 16;
  v8f acc = {};
#pragma unroll
  for (int kk = 0; kk < 4; ++kk) {
    v16h af = load_a(x + (size_t)r0 * Hc + kk * 32, Hc, 1, 32);
    v16h bf = load_b(sw + (size_t)c0 * Hc + kk * 32, 1, Hc, 32);
    acc = wmma32(af, bf, acc);
  }
  int lane = threadIdx.x & 31;
  int n = lane & 15, mb = (lane >> 4) * 8;
#pragma unroll
  for (int r = 0; r < 8; ++r) {
    int row = r0 + mb + r, col = c0 + n;
    out[(size_t)row * Sc + col] = acc[r] + sb[col];
  }
}

// reward = x @ rdec_w^T + rdec_b  (one dot of length 128 per row)
__global__ void k_rdec(const float* __restrict__ x, const float* __restrict__ rw,
                       const float* __restrict__ rb, float* __restrict__ out) {
  int row = blockIdx.x * blockDim.x + threadIdx.x;
  if (row >= ROWS) return;
  float acc = rb[0];
#pragma unroll 8
  for (int k = 0; k < Hc; ++k) acc += x[(size_t)row * Hc + k] * rw[k];
  out[row] = acc;
}

// ---------------------------------------------------------------------------

extern "C" void kernel_launch(void* const* d_in, const int* in_sizes, int n_in,
                              void* d_out, int out_size, void* d_ws, size_t ws_size,
                              hipStream_t stream) {
  (void)in_sizes; (void)n_in; (void)out_size; (void)ws_size;
  const float* st  = (const float*)d_in[0];
  const float* ac  = (const float*)d_in[1];
  const float* ew  = (const float*)d_in[2];
  const float* eb  = (const float*)d_in[3];
  const float* nw  = (const float*)d_in[4];
  const float* nb  = (const float*)d_in[5];
  const float* Ab  = (const float*)d_in[6];   // complex64 interleaved
  const float* Bb  = (const float*)d_in[7];
  const float* Cb  = (const float*)d_in[8];
  const float* Dp  = (const float*)d_in[9];
  const float* ow  = (const float*)d_in[10];
  const float* ob  = (const float*)d_in[11];
  const float* o2w = (const float*)d_in[12];
  const float* o2b = (const float*)d_in[13];
  const float* sw  = (const float*)d_in[14];
  const float* sb  = (const float*)d_in[15];
  const float* rw  = (const float*)d_in[16];
  const float* rb  = (const float*)d_in[17];
  float* out = (float*)d_out;

  float* ws = (float*)d_ws;
  size_t off = 0;
  auto alloc = [&](size_t n) { float* p = ws + off; off += n; return p; };
  float* Pa   = alloc((size_t)LH * RS * RS);   // 32 MB
  float* Pb   = alloc((size_t)LH * RS * RS);   // 32 MB
  float* Bh   = alloc((size_t)LH * RS);
  float* Chv  = alloc((size_t)LH * RS);
  float* What = alloc((size_t)LH * CH * RS);
  float* Mhat = alloc((size_t)LH * RS * CH);
  float* That = alloc((size_t)LH * CH * 32);
  float* xa   = alloc((size_t)ROWS * Hc);
  float* xb   = alloc((size_t)ROWS * Hc);
  float* xn   = alloc((size_t)ROWS * Hc);
  float* uT   = alloc((size_t)ROWS * Hc + 64);
  // Pa is dead after the squaring chain -> reuse for per-layer scratch:
  float* Hmat = Pa;                                    // 16 MB
  float* yg   = Pa + (size_t)Bc * Hc * RS * NC;        // 2 MB

  // --- weight preprocessing (deterministic, redone every call) ---
  k_encode<<<(ROWS * Hc + 255) / 256, 256, 0, stream>>>(st, ac, ew, eb, xa);
  k_embed<<<LH, 256, 0, stream>>>(Ab, Pb);             // Pb = R(A)
  k_bc<<<LH, RS, 0, stream>>>(Bb, Cb, Bh, Chv);
  k_wmk<<<LH, RS, 0, stream>>>(Pb, Bh, Chv, Dp, What, Mhat, That);
  k_sq<<<LH * 8, 256, 0, stream>>>(Pb, Pa);            // Pa = R^2
  k_sq<<<LH * 8, 256, 0, stream>>>(Pa, Pb);            // Pb = R^4
  k_sq<<<LH * 8, 256, 0, stream>>>(Pb, Pa);            // Pa = R^8
  k_sq<<<LH * 8, 256, 0, stream>>>(Pa, Pb);            // Pb = R^16 = P

  // --- layer pipeline ---
  size_t smem = (size_t)(RS * (RS + 1) + RS * (CH + 1) + RS + CH) * sizeof(float);
  float* cur = xa;
  float* nxt = xb;
  for (int l = 0; l < NLc; ++l) {
    k_ln<<<ROWS, Hc, 0, stream>>>(cur, nw, nb, l, xn, uT);
    k_scan<<<Bc * Hc, RS, smem, stream>>>(Pb, Mhat, xn, l, Hmat);
    k_out<<<Bc * Hc, 256, 0, stream>>>(What, That, Hmat, uT, l, yg);
    k_gate<<<ROWS / 16, 256, 0, stream>>>(yg, ow, ob, o2w, o2b, cur, l, nxt);
    float* t = cur; cur = nxt; nxt = t;
  }

  // --- decoders ---
  k_sdec<<<ROWS / 16, 64, 0, stream>>>(cur, sw, sb, out);
  k_rdec<<<(ROWS + 255) / 256, 256, 0, stream>>>(cur, rw, rb, out + (size_t)ROWS * Sc);
}
